// MsaAttentionBlock_39582418600373
// MI455X (gfx1250) — compile-verified, matrix operand order checked by reading, not run on hardware
//
#include <hip/hip_runtime.h>
#include <hip/hip_bf16.h>

// ---------------- problem constants ----------------
#define Bq 1
#define Sq 128
#define Nq 256
#define Dq 256
#define Hq 8
#define DHq 32
#define INNERq 256
#define Mrows (Sq * Nq)          // 32768 rows for both phases
#define NEGV (-1e9f)

typedef __attribute__((ext_vector_type(16))) _Float16 v16h;
typedef __attribute__((ext_vector_type(8)))  _Float16 v8h;
typedef __attribute__((ext_vector_type(8)))  float    v8f;

// Build a 16x32 f16 A/B fragment from two contiguous 8-half chunks.
// A-matrix 16-bit layout (ISA 7.12.2): lane group g = lane/16,
// chunk0 covers K = 8g..8g+7, chunk1 covers K = 16+8g..16+8g+7.
__device__ __forceinline__ v16h load_frag(const _Float16* p0, const _Float16* p1) {
    v8h a = *(const v8h*)p0;
    v8h b = *(const v8h*)p1;
    v16h r;
#pragma unroll
    for (int i = 0; i < 8; ++i) { r[i] = a[i]; r[8 + i] = b[i]; }
    return r;
}

__device__ __forceinline__ v8f wmma_f16(v16h a, v16h b, v8f c) {
    return __builtin_amdgcn_wmma_f32_16x16x32_f16(false, a, false, b, (short)0, c,
                                                  false, false);
}

// ---------------- weight pack: fp32 [K=256, ld] -> f16 WMMA-B tiles ----------------
// Output layout: tile(kb,nt) -> 512 halves: lane l holds 16 halves, half i maps to
// (k = kb*32 + pat(g,i), n = nt*16 + l%16) mirroring the A-matrix K pattern.
__global__ void pack_b_kernel(const float* __restrict__ W, _Float16* __restrict__ out,
                              int ld, int col_off) {
    int idx = blockIdx.x * 256 + threadIdx.x;          // K*Nout = 65536 halves
    if (idx >= 256 * 256) return;
    int i    = idx & 15;
    int l    = (idx >> 4) & 31;
    int tile = idx >> 9;
    const int NT = 16;                                 // Nout/16
    int nt = tile % NT;
    int kb = tile / NT;
    int g  = l >> 4;
    int base = (i < 8) ? (8 * g + i) : (16 + 8 * g + (i - 8));
    int k = kb * 32 + base;
    int n = nt * 16 + (l & 15);
    out[idx] = (_Float16)W[k * ld + col_off + n];
}

// ---------------- LayerNorm -> f16 ----------------
__global__ void ln_kernel(const float* __restrict__ src, const float* __restrict__ gam,
                          const float* __restrict__ bet, _Float16* __restrict__ dst,
                          int transpose, int Sdim, int Ndim) {
    int row = blockIdx.x;
    int t   = threadIdx.x;                             // 256 threads, one per D elem
    int srow = row;
    if (transpose) {                                   // dst row = n*S + s reads src (s,n)
        int n = row / Sdim, s = row % Sdim;
        srow = s * Ndim + n;
    }
    float v = src[(size_t)srow * Dq + t];
    __shared__ float red[256];
    red[t] = v; __syncthreads();
#pragma unroll
    for (int off = 128; off > 0; off >>= 1) {
        if (t < off) red[t] += red[t + off];
        __syncthreads();
    }
    float mu = red[0] * (1.0f / Dq);
    __syncthreads();
    float d = v - mu;
    red[t] = d * d; __syncthreads();
#pragma unroll
    for (int off = 128; off > 0; off >>= 1) {
        if (t < off) red[t] += red[t + off];
        __syncthreads();
    }
    float var = red[0] * (1.0f / Dq);
    float y = d * rsqrtf(var + 1e-5f) * gam[t] + bet[t];
    dst[(size_t)row * Dq + t] = (_Float16)y;
}

// ---------------- pairwise bias: bias[h,i,j] = sum_d pair[i,j,d] * we[d,h] ----------
__global__ void bias_kernel(const float* __restrict__ pair, const float* __restrict__ we,
                            float* __restrict__ bias) {
    int ij = blockIdx.x;                               // 0..N*N-1
    int t  = threadIdx.x;                              // 256
    __shared__ float row[256];
    __shared__ float part[256];
    row[t] = pair[(size_t)ij * Dq + t];
    __syncthreads();
    int h = t & 7, dg = t >> 3;                        // 32 d-groups x 8 heads
    float s = 0.f;
#pragma unroll 8
    for (int d = dg; d < 256; d += 32) s += row[d] * we[d * Hq + h];
    part[t] = s; __syncthreads();
    if (t < 8) {
        float acc = 0.f;
#pragma unroll
        for (int k = 0; k < 32; ++k) acc += part[k * 8 + h];
        bias[(size_t)h * (Nq * Nq) + ij] = acc;
    }
}

// ---------------- WMMA GEMM: 16x64 output tile per wave, K=256 ----------------
// Each wave owns 4 adjacent 16x16 N-tiles; the A fragment is loaded once per
// K-step and reused across 4 WMMAs (4x lower A traffic, 32 WMMAs per wave).
// epi: 0=f16 store   1=f16 transposed V store [(b*H+h)*DH+dh][pos]
//      2=f32 sigmoid(acc+bvec)   3=f32 residual+bias store (optional s/n remap)
__global__ void gemm_kernel(const _Float16* __restrict__ A, const _Float16* __restrict__ Bp,
                            int M, int Nout, int epi,
                            const float* __restrict__ bvec, const float* __restrict__ residual,
                            float* __restrict__ outF32, _Float16* __restrict__ outF16,
                            int L, int Sdim, int Ndim) {
    int wave = blockIdx.x * (blockDim.x >> 5) + (threadIdx.x >> 5);
    int lane = threadIdx.x & 31;
    int NT  = Nout >> 4;                               // 16x16 tiles along N
    int NG  = Nout >> 6;                               // 16x64 groups along N
    int mt = wave / NG, ng = wave % NG;
    int g = lane >> 4, ln = lane & 15;

    const _Float16* arow = A + (size_t)(mt * 16 + ln) * 256;
    v8f acc0 = {}, acc1 = {}, acc2 = {}, acc3 = {};
#pragma unroll
    for (int kb = 0; kb < 8; ++kb) {
        v16h a = load_frag(arow + kb * 32 + 8 * g, arow + kb * 32 + 16 + 8 * g);
        const _Float16* bb = Bp + (size_t)(kb * NT + ng * 4) * 512 + lane * 16;
        v16h b0 = *(const v16h*)(bb);
        v16h b1 = *(const v16h*)(bb + 512);
        v16h b2 = *(const v16h*)(bb + 1024);
        v16h b3 = *(const v16h*)(bb + 1536);
        acc0 = wmma_f16(a, b0, acc0);
        acc1 = wmma_f16(a, b1, acc1);
        acc2 = wmma_f16(a, b2, acc2);
        acc3 = wmma_f16(a, b3, acc3);
    }
    v8f accs[4] = {acc0, acc1, acc2, acc3};
#pragma unroll
    for (int t = 0; t < 4; ++t) {
        int col = (ng * 4 + t) * 16 + ln;
#pragma unroll
        for (int r = 0; r < 8; ++r) {
            int row = mt * 16 + r + 8 * g;
            float v = accs[t][r];
            if (epi == 0) {
                outF16[(size_t)row * Nout + col] = (_Float16)v;
            } else if (epi == 1) {
                int h = col >> 5, dh = col & 31;
                int ba = row / L, pos = row % L;
                outF16[(size_t)((ba * Hq + h) * DHq + dh) * L + pos] = (_Float16)v;
            } else if (epi == 2) {
                float z = v + bvec[col];
                outF32[(size_t)row * Nout + col] = 1.f / (1.f + __expf(-z));
            } else {
                float z = v + bvec[col];
                size_t idx;
                if (Sdim) { int nn = row / Sdim, ss = row % Sdim;
                            idx = (size_t)(ss * Ndim + nn) * Nout + col; }
                else      { idx = (size_t)row * Nout + col; }
                outF32[idx] = residual[idx] + z;
            }
        }
    }
}

// ---------------- flash attention: 1 wave = 16 query rows of one (batch, head) ------
__global__ void attn_kernel(const _Float16* __restrict__ Q, const _Float16* __restrict__ Kp,
                            const _Float16* __restrict__ VT, const float* __restrict__ bias,
                            const unsigned char* __restrict__ mask,
                            const float* __restrict__ gate, _Float16* __restrict__ Octx,
                            int L, int mbs, int mps) {
    int wslot = threadIdx.x >> 5, lane = threadIdx.x & 31;
    int w = blockIdx.x * (blockDim.x >> 5) + wslot;
    int qtiles = L >> 4;
    int qt = w % qtiles;
    int h  = (w / qtiles) & 7;
    int batch = w / (qtiles * Hq);
    int g = lane >> 4, ln = lane & 15;
    const float scale = 0.17677669529663687f;          // 1/sqrt(32)

    __shared__ __align__(32) _Float16 lds[8][16 * 32];
    _Float16* myP = lds[wslot];

    // Q fragment (A layout: this lane's A row is m = ln)
    const _Float16* qrow = Q + (size_t)(batch * L + qt * 16 + ln) * INNERq + h * DHq;
    v16h qa = load_frag(qrow + 8 * g, qrow + 16 + 8 * g);

    bool mq[8];
#pragma unroll
    for (int r = 0; r < 8; ++r) {                      // C-layout rows: m = r + 8g
        int qp = qt * 16 + r + 8 * g;
        mq[r] = mask[batch * mbs + qp * mps] != 0;
    }
    float mrun[8], lrun[8];
#pragma unroll
    for (int r = 0; r < 8; ++r) { mrun[r] = -INFINITY; lrun[r] = 0.f; }
    v8f o0 = {}, o1 = {};

    for (int kb = 0; kb < L; kb += 32) {
        int keyA = kb + ln, keyB = kb + 16 + ln;
        const _Float16* k0 = Kp + (size_t)(batch * L + keyA) * INNERq + h * DHq;
        const _Float16* k1 = Kp + (size_t)(batch * L + keyB) * INNERq + h * DHq;
        v16h kfa = load_frag(k0 + 8 * g, k0 + 16 + 8 * g);
        v16h kfb = load_frag(k1 + 8 * g, k1 + 16 + 8 * g);
        v8f z = {};
        v8f s0 = wmma_f16(qa, kfa, z);
        v8f s1 = wmma_f16(qa, kfb, z);

        bool mA = mask[batch * mbs + keyA * mps] != 0;
        bool mB = mask[batch * mbs + keyB * mps] != 0;

        float x0[8], x1[8];
#pragma unroll
        for (int r = 0; r < 8; ++r) {
            int qp = qt * 16 + r + 8 * g;
            float a = s0[r] * scale, b = s1[r] * scale;
            if (bias) {
                const float* bp = bias + ((size_t)h * L + qp) * L;
                a += bp[keyA]; b += bp[keyB];
            }
            x0[r] = (mq[r] && mA) ? a : NEGV;
            x1[r] = (mq[r] && mB) ? b : NEGV;
        }
        // online softmax; row = 16 lanes of same half-wave (wave32 -> xor within 16)
#pragma unroll
        for (int r = 0; r < 8; ++r) {
            float tm = fmaxf(x0[r], x1[r]);
#pragma unroll
            for (int off = 1; off < 16; off <<= 1) tm = fmaxf(tm, __shfl_xor(tm, off, 32));
            float nm   = fmaxf(mrun[r], tm);
            float corr = __expf(mrun[r] - nm);
            float p0 = __expf(x0[r] - nm), p1 = __expf(x1[r] - nm);
            float rs = p0 + p1;
#pragma unroll
            for (int off = 1; off < 16; off <<= 1) rs += __shfl_xor(rs, off, 32);
            lrun[r] = lrun[r] * corr + rs;
            mrun[r] = nm;
            o0[r] *= corr; o1[r] *= corr;
            myP[(r + 8 * g) * 32 + ln]      = (_Float16)p0;   // C layout -> LDS tile
            myP[(r + 8 * g) * 32 + 16 + ln] = (_Float16)p1;
        }
        __builtin_amdgcn_wave_barrier();
        // reload P in A layout (lane's row m = ln, K = key-within-block pattern)
        v16h pa = load_frag(&myP[ln * 32 + 8 * g], &myP[ln * 32 + 16 + 8 * g]);
        __builtin_amdgcn_wave_barrier();
        // V B-fragments: K-dim = key (contiguous via VT), N-dim = dh
        const _Float16* v0 = VT + ((size_t)(batch * Hq + h) * DHq + ln)      * L + kb;
        const _Float16* v1 = VT + ((size_t)(batch * Hq + h) * DHq + 16 + ln) * L + kb;
        v16h vb0 = load_frag(v0 + 8 * g, v0 + 16 + 8 * g);
        v16h vb1 = load_frag(v1 + 8 * g, v1 + 16 + 8 * g);
        o0 = wmma_f16(pa, vb0, o0);
        o1 = wmma_f16(pa, vb1, o1);
    }
#pragma unroll
    for (int r = 0; r < 8; ++r) {
        float inv = 1.f / lrun[r];
        int qp = qt * 16 + r + 8 * g;
        size_t base = (size_t)(batch * L + qp) * INNERq + h * DHq;
        float g0 = gate[base + ln], g1 = gate[base + 16 + ln];
        Octx[base + ln]      = (_Float16)(o0[r] * inv * g0);
        Octx[base + 16 + ln] = (_Float16)(o1[r] * inv * g1);
    }
}

// ---------------- host side ----------------
extern "C" void kernel_launch(void* const* d_in, const int* in_sizes, int n_in,
                              void* d_out, int out_size, void* d_ws, size_t ws_size,
                              hipStream_t stream) {
    (void)in_sizes; (void)n_in; (void)out_size; (void)ws_size;

    const float* x      = (const float*)d_in[0];
    const float* pair   = (const float*)d_in[1];
    const unsigned char* mask = (const unsigned char*)d_in[2];
    const float* row_ln_g = (const float*)d_in[3];
    const float* row_ln_b = (const float*)d_in[4];
    const float* row_wq   = (const float*)d_in[5];
    const float* row_wkv  = (const float*)d_in[6];
    const float* row_wg   = (const float*)d_in[7];
    const float* row_bg   = (const float*)d_in[8];
    const float* row_wo   = (const float*)d_in[9];
    const float* row_bo   = (const float*)d_in[10];
    const float* row_we   = (const float*)d_in[11];
    const float* col_ln_g = (const float*)d_in[12];
    const float* col_ln_b = (const float*)d_in[13];
    const float* col_wq   = (const float*)d_in[14];
    const float* col_wkv  = (const float*)d_in[15];
    const float* col_wg   = (const float*)d_in[16];
    const float* col_bg   = (const float*)d_in[17];
    const float* col_wo   = (const float*)d_in[18];
    const float* col_bo   = (const float*)d_in[19];
    float* out = (float*)d_out;

    // workspace carve-up
    char* ws = (char*)d_ws;
    const size_t f16Mat = (size_t)Mrows * INNERq * sizeof(_Float16);   // 16.78 MB
    _Float16* xn   = (_Float16*)(ws);                 ws += f16Mat;
    _Float16* qb   = (_Float16*)(ws);                 ws += f16Mat;
    _Float16* kb_  = (_Float16*)(ws);                 ws += f16Mat;
    _Float16* vT   = (_Float16*)(ws);                 ws += f16Mat;
    _Float16* ctx  = (_Float16*)(ws);                 ws += f16Mat;
    float*    gate = (float*)(ws);                    ws += (size_t)Mrows * INNERq * sizeof(float);
    float*    bias = (float*)(ws);                    ws += (size_t)Hq * Nq * Nq * sizeof(float);
    const size_t wHalves = 256 * 256;
    _Float16* wq_r = (_Float16*)(ws); ws += wHalves * 2;
    _Float16* wk_r = (_Float16*)(ws); ws += wHalves * 2;
    _Float16* wv_r = (_Float16*)(ws); ws += wHalves * 2;
    _Float16* wg_r = (_Float16*)(ws); ws += wHalves * 2;
    _Float16* wo_r = (_Float16*)(ws); ws += wHalves * 2;
    _Float16* wq_c = (_Float16*)(ws); ws += wHalves * 2;
    _Float16* wk_c = (_Float16*)(ws); ws += wHalves * 2;
    _Float16* wv_c = (_Float16*)(ws); ws += wHalves * 2;
    _Float16* wg_c = (_Float16*)(ws); ws += wHalves * 2;
    _Float16* wo_c = (_Float16*)(ws); ws += wHalves * 2;

    dim3 blk(256);
    dim3 packGrid(256);                                // 65536/256
    // ---- weight packing (f32 -> f16, WMMA B layout) ----
    pack_b_kernel<<<packGrid, blk, 0, stream>>>(row_wq,  wq_r, 256, 0);
    pack_b_kernel<<<packGrid, blk, 0, stream>>>(row_wkv, wk_r, 512, 0);
    pack_b_kernel<<<packGrid, blk, 0, stream>>>(row_wkv, wv_r, 512, 256);
    pack_b_kernel<<<packGrid, blk, 0, stream>>>(row_wg,  wg_r, 256, 0);
    pack_b_kernel<<<packGrid, blk, 0, stream>>>(row_wo,  wo_r, 256, 0);
    pack_b_kernel<<<packGrid, blk, 0, stream>>>(col_wq,  wq_c, 256, 0);
    pack_b_kernel<<<packGrid, blk, 0, stream>>>(col_wkv, wk_c, 512, 0);
    pack_b_kernel<<<packGrid, blk, 0, stream>>>(col_wkv, wv_c, 512, 256);
    pack_b_kernel<<<packGrid, blk, 0, stream>>>(col_wg,  wg_c, 256, 0);
    pack_b_kernel<<<packGrid, blk, 0, stream>>>(col_wo,  wo_c, 256, 0);

    dim3 gemmGrid(1024);                               // 8192 waves (16x64 tiles) / 8
    dim3 lnGrid(Mrows);
    dim3 biasGrid(Nq * Nq);

    // ================= row attention =================
    ln_kernel<<<lnGrid, blk, 0, stream>>>(x, row_ln_g, row_ln_b, xn, 0, 0, 0);
    bias_kernel<<<biasGrid, blk, 0, stream>>>(pair, row_we, bias);
    gemm_kernel<<<gemmGrid, blk, 0, stream>>>(xn, wq_r, Mrows, 256, 0, nullptr, nullptr,
                                              nullptr, qb, 0, 0, 0);
    gemm_kernel<<<gemmGrid, blk, 0, stream>>>(xn, wk_r, Mrows, 256, 0, nullptr, nullptr,
                                              nullptr, kb_, 0, 0, 0);
    gemm_kernel<<<gemmGrid, blk, 0, stream>>>(xn, wv_r, Mrows, 256, 1, nullptr, nullptr,
                                              nullptr, vT, 256, 0, 0);
    gemm_kernel<<<gemmGrid, blk, 0, stream>>>(xn, wg_r, Mrows, 256, 2, row_bg, nullptr,
                                              gate, nullptr, 0, 0, 0);
    // 128 batches * 8 heads * 16 qtiles = 16384 waves -> 2048 blocks
    attn_kernel<<<dim3(2048), blk, 0, stream>>>(qb, kb_, vT, bias, mask, gate, ctx,
                                                256, 256, 1);
    gemm_kernel<<<gemmGrid, blk, 0, stream>>>(ctx, wo_r, Mrows, 256, 3, row_bo, x,
                                              out, nullptr, 0, 0, 0);

    // ================= column attention =================
    ln_kernel<<<lnGrid, blk, 0, stream>>>(out, col_ln_g, col_ln_b, xn, 1, Sq, Nq);
    gemm_kernel<<<gemmGrid, blk, 0, stream>>>(xn, wq_c, Mrows, 256, 0, nullptr, nullptr,
                                              nullptr, qb, 0, 0, 0);
    gemm_kernel<<<gemmGrid, blk, 0, stream>>>(xn, wk_c, Mrows, 256, 0, nullptr, nullptr,
                                              nullptr, kb_, 0, 0, 0);
    gemm_kernel<<<gemmGrid, blk, 0, stream>>>(xn, wv_c, Mrows, 256, 1, nullptr, nullptr,
                                              nullptr, vT, 128, 0, 0);
    gemm_kernel<<<gemmGrid, blk, 0, stream>>>(xn, wg_c, Mrows, 256, 2, col_bg, nullptr,
                                              gate, nullptr, 0, 0, 0);
    // 256 batches * 8 heads * 8 qtiles = 16384 waves -> 2048 blocks
    attn_kernel<<<dim3(2048), blk, 0, stream>>>(qb, kb_, vT, nullptr, mask, gate, ctx,
                                                128, 1, 256);
    gemm_kernel<<<gemmGrid, blk, 0, stream>>>(ctx, wo_c, Mrows, 256, 3, col_bo, out,
                                              out, nullptr, 0, Sq, Nq);
}
